// MatchingModule_6382321401917
// MI455X (gfx1250) — matrix operand order repaired and checked by reading, not compiled, not run on hardware
//
#include <hip/hip_runtime.h>

// ---------------------------------------------------------------------------
// MatchingModule for MI455X (gfx1250, wave32, WMMA 16x16x32 f16 -> f32 accum)
// ---------------------------------------------------------------------------

typedef __attribute__((ext_vector_type(16))) _Float16 v16h;
typedef __attribute__((ext_vector_type(8)))  float    v8f;
typedef int v4i_vs __attribute__((vector_size(16)));   // type expected by async-LDS builtin

#define B_ 8
#define C_ 128
#define H_ 128
#define W_ 128

// padded geometries (channels-last f16 buffers)
#define F2P_H 134   // warp result, halo 3 (for 7x7 correlation window)
#define F2P_W 134
#define C1P_H 130   // corr / h1 buffers, halo 1 (for 3x3 convs)
#define C1P_W 130
#define H2P_H 132   // h2 buffer, halo 2 (for 5x5 conv)
#define H2P_W 132

// Async memory -> LDS path (CDNA5 ASYNCcnt). Guarded so the file compiles on
// toolchains where the builtins are absent (falls back to sync LDS fill).
#if defined(__gfx1250__) && \
    __has_builtin(__builtin_amdgcn_global_load_async_to_lds_b128) && \
    __has_builtin(__builtin_amdgcn_s_wait_asynccnt)
#define HAVE_ASYNC_LDS 1
#else
#define HAVE_ASYNC_LDS 0
#endif

static __device__ __forceinline__ unsigned short f2h_bits(float f) {
    union { _Float16 h; unsigned short u; } c; c.h = (_Float16)f; return c.u;
}

// ---------------------------------------------------------------------------
// 1) bilinear 2x upsample of flow (align_corners=False / half-pixel centers)
// ---------------------------------------------------------------------------
__global__ void upsample_kernel(const float* __restrict__ flow, float* __restrict__ flow_up) {
    int t = blockIdx.x * blockDim.x + threadIdx.x;
    if (t >= B_ * 2 * H_ * W_) return;
    int x = t & 127, y = (t >> 7) & 127, bc = t >> 14;
    float sx = 0.5f * x - 0.25f;
    float sy = 0.5f * y - 0.25f;
    float x0f = floorf(sx), y0f = floorf(sy);
    int ix0 = (int)x0f, iy0 = (int)y0f;
    float wx = sx - x0f, wy = sy - y0f;
    int x0c = min(max(ix0, 0), 63),     x1c = min(max(ix0 + 1, 0), 63);
    int y0c = min(max(iy0, 0), 63),     y1c = min(max(iy0 + 1, 0), 63);
    const float* src = flow + (size_t)bc * 64 * 64;
    float v00 = src[y0c * 64 + x0c], v01 = src[y0c * 64 + x1c];
    float v10 = src[y1c * 64 + x0c], v11 = src[y1c * 64 + x1c];
    flow_up[t] = v00 * (1.f - wx) * (1.f - wy) + v01 * wx * (1.f - wy)
               + v10 * (1.f - wx) * wy         + v11 * wx * wy;
}

// ---------------------------------------------------------------------------
// 2) features1: f32 NCHW -> f16 NHWC (channel-contiguous for correlation)
// ---------------------------------------------------------------------------
__global__ void pack_f1_kernel(const float* __restrict__ f1, unsigned short* __restrict__ f1cl) {
    long long t = (long long)blockIdx.x * blockDim.x + threadIdx.x;
    if (t >= (long long)B_ * H_ * W_ * C_) return;
    int c = (int)(t & 127);
    int x = (int)((t >> 7) & 127);
    int y = (int)((t >> 14) & 127);
    int b = (int)(t >> 21);
    float v = f1[((size_t)(b * C_ + c) * H_ + y) * W_ + x];
    f1cl[t] = f2h_bits(v);
}

// ---------------------------------------------------------------------------
// 3) backward warp of features2 by 2.5*flow_up, f16 NHWC with halo 3
//    (halo stays zero thanks to the memset on f2wp)
// ---------------------------------------------------------------------------
__global__ void warp_kernel(const float* __restrict__ f2, const float* __restrict__ flow_up,
                            unsigned short* __restrict__ f2wp) {
    int t = blockIdx.x * blockDim.x + threadIdx.x;     // B*H*W*(C/8)
    if (t >= B_ * H_ * W_ * (C_ / 8)) return;
    int cg = t & 15;
    int x  = (t >> 4) & 127;
    int y  = (t >> 11) & 127;
    int b  = t >> 18;
    float fx = flow_up[((size_t)(b * 2 + 0) * H_ + y) * W_ + x] * 2.5f;
    float fy = flow_up[((size_t)(b * 2 + 1) * H_ + y) * W_ + x] * 2.5f;
    float sx = (float)x + fx, sy = (float)y + fy;
    float x0f = floorf(sx), y0f = floorf(sy);
    int ix0 = (int)x0f, iy0 = (int)y0f;
    float wx = sx - x0f, wy = sy - y0f;
    float w00 = (1.f - wx) * (1.f - wy), w01 = wx * (1.f - wy);
    float w10 = (1.f - wx) * wy,         w11 = wx * wy;
    bool vx0 = (ix0 >= 0) && (ix0 < W_);
    bool vx1 = (ix0 + 1 >= 0) && (ix0 + 1 < W_);
    bool vy0 = (iy0 >= 0) && (iy0 < H_);
    bool vy1 = (iy0 + 1 >= 0) && (iy0 + 1 < H_);
    int xc0 = min(max(ix0, 0), W_ - 1), xc1 = min(max(ix0 + 1, 0), W_ - 1);
    int yc0 = min(max(iy0, 0), H_ - 1), yc1 = min(max(iy0 + 1, 0), H_ - 1);
    unsigned short* dst = f2wp + ((size_t)(b * F2P_H + (y + 3)) * F2P_W + (x + 3)) * C_ + cg * 8;
    #pragma unroll
    for (int i = 0; i < 8; ++i) {
        int c = cg * 8 + i;
        const float* sp = f2 + (size_t)(b * C_ + c) * H_ * W_;
        float v = 0.f;
        if (vy0 && vx0) v += w00 * sp[yc0 * W_ + xc0];
        if (vy0 && vx1) v += w01 * sp[yc0 * W_ + xc1];
        if (vy1 && vx0) v += w10 * sp[yc1 * W_ + xc0];
        if (vy1 && vx1) v += w11 * sp[yc1 * W_ + xc1];
        dst[i] = f2h_bits(v);
    }
}

// ---------------------------------------------------------------------------
// 4) 7x7 cost volume over C=128 -> corr, channels-last padded to 64 ch, halo 1
//    16x16 pixel tile / block; 22x22 x 32ch f2w tile staged in LDS.
//    LDS fill uses GLOBAL_LOAD_ASYNC_TO_LDS_B128 (ASYNCcnt) when available.
// ---------------------------------------------------------------------------
__global__ void corr_kernel(const unsigned short* __restrict__ f1cl,
                            const unsigned short* __restrict__ f2wp,
                            unsigned short* __restrict__ corrp) {
    __shared__ unsigned int s_f2[484 * 16];            // 22*22 px * 16 dwords (32 ch) = 31KB
    int tid = threadIdx.x;
    int lx = tid & 15, ly = tid >> 4;
    int blk = blockIdx.x;                              // b*64 + ty*8 + tx
    int tx = blk & 7, ty = (blk >> 3) & 7, b = blk >> 6;
    int x0 = tx * 16, y0 = ty * 16;
    int gx = x0 + lx, gy = y0 + ly;

    float acc[49];
    #pragma unroll
    for (int d = 0; d < 49; ++d) acc[d] = 0.f;

    const unsigned short* f1p = f1cl + ((size_t)(b * H_ + gy) * W_ + gx) * C_;
    int base = (ly * 22 + lx) * 16;

    for (int ch = 0; ch < 4; ++ch) {                   // 4 chunks of 32 channels
        __syncthreads();
#if HAVE_ASYNC_LDS
        // each iteration moves 16B memory->LDS without touching VGPR data
        for (int i = tid; i < 484 * 4; i += 256) {
            int pix = i >> 2, q = i & 3;               // q: 16B quarter of 64B/px chunk
            int qy = pix / 22;
            int qx = pix - qy * 22;
            const unsigned short* src =
                f2wp + ((size_t)(b * F2P_H + (y0 + qy)) * F2P_W + (x0 + qx)) * C_ + ch * 32 + q * 8;
            __builtin_amdgcn_global_load_async_to_lds_b128(
                (__attribute__((address_space(1))) v4i_vs*)src,
                (__attribute__((address_space(3))) v4i_vs*)&s_f2[pix * 16 + q * 4],
                0, 0);
        }
        __builtin_amdgcn_s_wait_asynccnt(0);
#else
        for (int i = tid; i < 484 * 16; i += 256) {
            int pix = i >> 4, dw = i & 15;
            int qy = pix / 22;
            int qx = pix - qy * 22;
            const unsigned int* src = (const unsigned int*)
                (f2wp + ((size_t)(b * F2P_H + (y0 + qy)) * F2P_W + (x0 + qx)) * C_ + ch * 32);
            s_f2[i] = src[dw];
        }
#endif
        __syncthreads();

        unsigned int a[16];
        const unsigned int* f1d = (const unsigned int*)(f1p + ch * 32);
        #pragma unroll
        for (int i = 0; i < 16; ++i) a[i] = f1d[i];

        #pragma unroll 4
        for (int dw = 0; dw < 16; ++dw) {
            union { unsigned int u; _Float16 h[2]; } av; av.u = a[dw];
            float a0 = (float)av.h[0], a1 = (float)av.h[1];
            #pragma unroll
            for (int d = 0; d < 49; ++d) {
                int dy = d / 7, dx = d - 7 * (d / 7);
                union { unsigned int u; _Float16 h[2]; } bv;
                bv.u = s_f2[base + (dy * 22 + dx) * 16 + dw];
                acc[d] += a0 * (float)bv.h[0] + a1 * (float)bv.h[1];
            }
        }
    }

    unsigned short* op = corrp + ((size_t)(b * C1P_H + (gy + 1)) * C1P_W + (gx + 1)) * 64;
    #pragma unroll
    for (int d = 0; d < 49; ++d) op[d] = f2h_bits(acc[d] * (1.f / 128.f));
    // channels 49..63 stay zero from the memset
}

// ---------------------------------------------------------------------------
// 5) Weight packing into per-lane WMMA B-fragment order.
//    A-layout (ISA table, 16-bit A 16x32): lane<16 -> K{0..7,16..23}, lane>=16 -> +8.
//    B-layout assumed (dense analog of documented B striping):
//      lane = n (0..15 per half-wave), VGPR v half h -> s = 2v+h, k = s (<8) or s+8 (>=8), +8 for lanes>=16.
// ---------------------------------------------------------------------------
__global__ void pack_weights_kernel(const float* __restrict__ w, unsigned int* __restrict__ dst,
                                    int O, int I, int KH, int KW, int KC, int NT) {
    int t = blockIdx.x * blockDim.x + threadIdx.x;
    int total = KH * KW * KC * NT * 32;
    if (t >= total) return;
    int lane = t & 31;
    int f = t >> 5;
    int nt = f % NT;
    int kc = (f / NT) % KC;
    int kw = (f / (NT * KC)) % KW;
    int kh =  f / (NT * KC * KW);
    int oc  = nt * 16 + (lane & 15);
    int khi = (lane >= 16) ? 8 : 0;
    unsigned int* out = dst + (size_t)f * 256 + lane * 8;
    #pragma unroll
    for (int v = 0; v < 8; ++v) {
        unsigned int d = 0;
        #pragma unroll
        for (int h = 0; h < 2; ++h) {
            int k  = 2 * v + h + ((v >= 4) ? 8 : 0) + khi;
            int ci = kc * 32 + k;
            float val = (ci < I && oc < O)
                ? w[((size_t)(oc * I + ci) * KH + kh) * KW + kw] : 0.f;
            d |= ((unsigned int)f2h_bits(val)) << (16 * h);
        }
        out[v] = d;
    }
}

// ---------------------------------------------------------------------------
// WMMA implicit-GEMM conv mainloop.
//   M = 16 output pixels (x0..x0+15, fixed y), N = NT*16 output channels,
//   K stepped 32 channels/tap. Input is channels-last f16, spatially padded,
//   so A-fragment = two contiguous 16B loads per lane (exact ISA A layout).
// ---------------------------------------------------------------------------
template <int KH, int KW, int KC, int NT, int INPH, int INPW, int INCH>
__device__ __forceinline__ void conv_mainloop(const unsigned short* __restrict__ inp,
                                              const uint4* __restrict__ wp,
                                              int b, int y, int x0, int lane, v8f* acc) {
    const int m    = lane & 15;
    const int koff = (lane >= 16) ? 8 : 0;
    int frag = 0;
    for (int kh = 0; kh < KH; ++kh) {
        // prefetch next tap row (global_prefetch_b8)
        if (kh + 1 < KH)
            __builtin_prefetch(inp + ((size_t)(b * INPH + (y + kh + 1)) * INPW + (x0 + m)) * INCH,
                               0, 1);
        for (int kw = 0; kw < KW; ++kw) {
            const unsigned short* px =
                inp + ((size_t)(b * INPH + (y + kh)) * INPW + (x0 + m + kw)) * INCH;
            #pragma unroll
            for (int kc = 0; kc < KC; ++kc) {
                union { v16h v; uint4 q[2]; } a;
                const unsigned short* ap = px + kc * 32 + koff;
                a.q[0] = *(const uint4*)(ap);
                a.q[1] = *(const uint4*)(ap + 16);
                #pragma unroll
                for (int nt = 0; nt < NT; ++nt) {
                    union { v16h v; uint4 q[2]; } bm;
                    const uint4* bp = wp + (size_t)(frag + nt) * 64 + lane * 2;
                    bm.q[0] = bp[0];
                    bm.q[1] = bp[1];
                    acc[nt] = __builtin_amdgcn_wmma_f32_16x16x32_f16(
                        false, a.v, false, bm.v, (short)0, acc[nt], false, false);
                }
                frag += NT;
            }
        }
    }
}

__device__ __forceinline__ void wave_tile(int* b, int* y, int* x0) {
    int wave = blockIdx.x * (blockDim.x >> 5) + (threadIdx.x >> 5);
    *x0 = (wave & 7) * 16;
    *y  = (wave >> 3) & 127;
    *b  = wave >> 10;
}

// conv1: corr(49 pad 64) -> 64ch, 3x3, lrelu; out h1p (halo 1)
__global__ void conv1_kernel(const unsigned short* __restrict__ corrp,
                             const uint4* __restrict__ wp1,
                             const float* __restrict__ bias1,
                             unsigned short* __restrict__ h1p) {
    int b, y, x0; wave_tile(&b, &y, &x0);
    int lane = threadIdx.x & 31;
    v8f acc[4] = {};
    conv_mainloop<3, 3, 2, 4, C1P_H, C1P_W, 64>(corrp, wp1, b, y, x0, lane, acc);
    int n  = lane & 15;
    int mh = (lane >= 16) ? 8 : 0;
    #pragma unroll
    for (int nt = 0; nt < 4; ++nt) {
        float bv = bias1[nt * 16 + n];
        #pragma unroll
        for (int r = 0; r < 8; ++r) {
            float s = acc[nt][r] + bv;
            s = (s >= 0.f) ? s : 0.1f * s;
            int m = r + mh;
            h1p[((size_t)(b * C1P_H + (y + 1)) * C1P_W + (x0 + m + 1)) * 64 + nt * 16 + n]
                = f2h_bits(s);
        }
    }
}

// conv2: 64 -> 32, 3x3, lrelu; out h2p (halo 2)
__global__ void conv2_kernel(const unsigned short* __restrict__ h1p,
                             const uint4* __restrict__ wp2,
                             const float* __restrict__ bias2,
                             unsigned short* __restrict__ h2p) {
    int b, y, x0; wave_tile(&b, &y, &x0);
    int lane = threadIdx.x & 31;
    v8f acc[2] = {};
    conv_mainloop<3, 3, 2, 2, C1P_H, C1P_W, 64>(h1p, wp2, b, y, x0, lane, acc);
    int n  = lane & 15;
    int mh = (lane >= 16) ? 8 : 0;
    #pragma unroll
    for (int nt = 0; nt < 2; ++nt) {
        float bv = bias2[nt * 16 + n];
        #pragma unroll
        for (int r = 0; r < 8; ++r) {
            float s = acc[nt][r] + bv;
            s = (s >= 0.f) ? s : 0.1f * s;
            int m = r + mh;
            h2p[((size_t)(b * H2P_H + (y + 2)) * H2P_W + (x0 + m + 2)) * 32 + nt * 16 + n]
                = f2h_bits(s);
        }
    }
}

// conv3: 32 -> 2, 5x5; out = flow_up + h (f32)
__global__ void conv3_kernel(const unsigned short* __restrict__ h2p,
                             const uint4* __restrict__ wp3,
                             const float* __restrict__ bias3,
                             const float* __restrict__ flow_up,
                             float* __restrict__ out) {
    int b, y, x0; wave_tile(&b, &y, &x0);
    int lane = threadIdx.x & 31;
    v8f acc[1] = {};
    conv_mainloop<5, 5, 1, 1, H2P_H, H2P_W, 32>(h2p, wp3, b, y, x0, lane, acc);
    int n  = lane & 15;
    int mh = (lane >= 16) ? 8 : 0;
    if (n < 2) {
        float bv = bias3[n];
        #pragma unroll
        for (int r = 0; r < 8; ++r) {
            int m = r + mh;
            size_t o = ((size_t)(b * 2 + n) * H_ + y) * W_ + (x0 + m);
            out[o] = acc[0][r] + bv + flow_up[o];
        }
    }
}

// ---------------------------------------------------------------------------
// launch
// ---------------------------------------------------------------------------
extern "C" void kernel_launch(void* const* d_in, const int* in_sizes, int n_in,
                              void* d_out, int out_size, void* d_ws, size_t ws_size,
                              hipStream_t stream) {
    const float* features1 = (const float*)d_in[0];
    const float* features2 = (const float*)d_in[1];
    const float* flow      = (const float*)d_in[2];
    const float* w1 = (const float*)d_in[3];
    const float* b1 = (const float*)d_in[4];
    const float* w2 = (const float*)d_in[5];
    const float* b2 = (const float*)d_in[6];
    const float* w3 = (const float*)d_in[7];
    const float* b3 = (const float*)d_in[8];
    float* out = (float*)d_out;
    (void)in_sizes; (void)n_in; (void)out_size; (void)ws_size;

    char* ws = (char*)d_ws;
    size_t off = 0;
    auto carve = [&](size_t bytes) -> char* {
        char* p = ws + off;
        off = (off + bytes + 255) & ~(size_t)255;
        return p;
    };
    const size_t flowup_b = (size_t)B_ * 2 * H_ * W_ * sizeof(float);
    const size_t f1cl_b   = (size_t)B_ * H_ * W_ * C_ * 2;
    const size_t f2wp_b   = (size_t)B_ * F2P_H * F2P_W * C_ * 2;
    const size_t corrp_b  = (size_t)B_ * C1P_H * C1P_W * 64 * 2;
    const size_t h1p_b    = (size_t)B_ * C1P_H * C1P_W * 64 * 2;
    const size_t h2p_b    = (size_t)B_ * H2P_H * H2P_W * 32 * 2;

    float*          flow_up = (float*)carve(flowup_b);
    unsigned short* f1cl    = (unsigned short*)carve(f1cl_b);
    unsigned short* f2wp    = (unsigned short*)carve(f2wp_b);
    unsigned short* corrp   = (unsigned short*)carve(corrp_b);
    unsigned short* h1p     = (unsigned short*)carve(h1p_b);
    unsigned short* h2p     = (unsigned short*)carve(h2p_b);
    unsigned int*   wp1     = (unsigned int*)carve(72 * 1024);
    unsigned int*   wp2     = (unsigned int*)carve(36 * 1024);
    unsigned int*   wp3     = (unsigned int*)carve(25 * 1024);

    // zero padded buffers: halos + corr channel padding come for free
    (void)hipMemsetAsync(f2wp,  0, f2wp_b,  stream);
    (void)hipMemsetAsync(corrp, 0, corrp_b, stream);
    (void)hipMemsetAsync(h1p,   0, h1p_b,   stream);
    (void)hipMemsetAsync(h2p,   0, h2p_b,   stream);

    upsample_kernel<<<(B_ * 2 * H_ * W_ + 255) / 256, 256, 0, stream>>>(flow, flow_up);
    pack_f1_kernel<<<(int)(((long long)B_ * H_ * W_ * C_ + 255) / 256), 256, 0, stream>>>(features1, f1cl);
    warp_kernel<<<(B_ * H_ * W_ * (C_ / 8) + 255) / 256, 256, 0, stream>>>(features2, flow_up, f2wp);

    pack_weights_kernel<<<(72 * 32 + 255) / 256, 256, 0, stream>>>(w1, wp1, 64, 49, 3, 3, 2, 4);
    pack_weights_kernel<<<(36 * 32 + 255) / 256, 256, 0, stream>>>(w2, wp2, 32, 64, 3, 3, 2, 2);
    pack_weights_kernel<<<(25 * 32 + 255) / 256, 256, 0, stream>>>(w3, wp3, 2, 32, 5, 5, 1, 1);

    corr_kernel<<<B_ * 8 * 8, 256, 0, stream>>>(f1cl, f2wp, corrp);

    // 8192 waves = B * H * (W/16); 4 waves per 128-thread block
    conv1_kernel<<<2048, 128, 0, stream>>>(corrp, (const uint4*)wp1, b1, h1p);
    conv2_kernel<<<2048, 128, 0, stream>>>(h1p,   (const uint4*)wp2, b2, h2p);
    conv3_kernel<<<2048, 128, 0, stream>>>(h2p,   (const uint4*)wp3, b3, flow_up, out);
}
